// FraudGCN_12335146074241
// MI455X (gfx1250) — matrix operand order repaired and checked by reading, not compiled, not run on hardware
//
#include <hip/hip_runtime.h>
#include <hip/hip_bf16.h>

// ---------------------------------------------------------------------------
// FraudGCN on MI455X (gfx1250).
//  - All dense matmuls on v_wmma_f32_16x16x32_f16 (wave32 WMMA).
//  - Weights pre-packed into the ISA B-fragment layout (f16) once per call,
//    so the GEMM inner loop is pure b128 vector loads + wmma.
//  - Activations carried in row-major f16 between layers (A fragments load as
//    two aligned v8h/b128 per lane per k-chunk, zero VALU converts in-loop).
//  - Graph aggregation via fp32 atomics; BatchNorm via LDS-reduced stats.
// ---------------------------------------------------------------------------

#define N_NODES 50000
#define N_EDGES 800000
#define MTILES  (N_NODES / 16)   // 3125 row tiles (exact)
#define ETILES  (N_EDGES / 16)   // 50000 edge tiles (exact)
#define BN_EPS  1e-5f

typedef __attribute__((ext_vector_type(16))) _Float16 v16h;
typedef __attribute__((ext_vector_type(8)))  _Float16 v8h;
typedef __attribute__((ext_vector_type(8)))  float    v8f;

// ---------------- fragment assembly helpers --------------------------------
__device__ inline v16h cat8(v8h lo, v8h hi) {
  return __builtin_shufflevector(lo, hi, 0, 1, 2, 3, 4, 5, 6, 7,
                                 8, 9, 10, 11, 12, 13, 14, 15);
}

// A fragment from row-major f16 tile (row = lane&15).  ISA 7.12.2:
//  lanes 0-15 take k = 0..7 / 16..23 of the chunk, lanes 16-31 take 8..15 / 24..31.
__device__ inline v16h load_a_frag_h(const _Float16* __restrict__ row,
                                     int k0, int khalf) {
  const v8h lo = *(const v8h*)(row + k0 + khalf * 8);
  const v8h hi = *(const v8h*)(row + k0 + 16 + khalf * 8);
  return cat8(lo, hi);
}

// ---------------- weight pre-pack into B-fragment layout --------------------
// For W[K,N] fp32: fragment (kt,nt) of 32 lanes; lane holds col n = nt*16+(lane&15),
// k = kt*32 + (lane>=16 ? 8 : 0) + 2v (+8 for v>=4), and +1, per VGPR v.  16 halves contiguous.
__global__ void pack_b_kernel(const float* __restrict__ W,
                              _Float16* __restrict__ Wp, int K, int N) {
  const int NT = N / 16;
  const int total = (K / 32) * NT * 32;
  for (int idx = blockIdx.x * blockDim.x + threadIdx.x; idx < total;
       idx += gridDim.x * blockDim.x) {
    const int lane  = idx & 31;
    const int frag  = idx >> 5;
    const int kt    = frag / NT;
    const int nt    = frag - kt * NT;
    const int n     = nt * 16 + (lane & 15);
    const int khalf = lane >> 4;
    _Float16* dst = Wp + (size_t)idx * 16;
#pragma unroll
    for (int v = 0; v < 8; ++v) {
      const int k = kt * 32 + khalf * 8 + 2 * v + (v >= 4 ? 8 : 0);
      dst[2 * v + 0] = (_Float16)W[(size_t)(k + 0) * N + n];
      dst[2 * v + 1] = (_Float16)W[(size_t)(k + 1) * N + n];
    }
  }
}

// ---------------- Dense GEMM: H[M,N] = Xh[M,K] @ Wp (f16 in, f32 out) ------
template <int K, int N>
__global__ void gemm_wmma_kernel(const _Float16* __restrict__ Xh,
                                 const _Float16* __restrict__ Wp,
                                 float* __restrict__ H, int mtiles) {
  const int wave = blockIdx.x * (blockDim.x >> 5) + (threadIdx.x >> 5);
  if (wave >= mtiles) return;               // wave-uniform: EXEC stays all-ones
  const int lane  = threadIdx.x & 31;
  const int khalf = lane >> 4;
  constexpr int NT = N / 16;
  constexpr int KT = K / 32;

  v8f acc[NT] = {};
  const _Float16* Xrow = Xh + (size_t)wave * 16 * K + (size_t)(lane & 15) * K;

#pragma unroll
  for (int kt = 0; kt < KT; ++kt) {
    const v16h a = load_a_frag_h(Xrow, kt * 32, khalf);
#pragma unroll
    for (int nt = 0; nt < NT; ++nt) {
      const v16h b = *(const v16h*)(Wp + ((size_t)(kt * NT + nt) * 32 + lane) * 16);
      acc[nt] = __builtin_amdgcn_wmma_f32_16x16x32_f16(
          false, a, false, b, (short)0, acc[nt], false, false);
    }
  }

  // C layout: VGPR r -> row (r + 8*(lane>=16)), col = lane&15
  float* Ht = H + (size_t)wave * 16 * N;
  const int mbase = khalf * 8;
  const int col   = lane & 15;
#pragma unroll
  for (int nt = 0; nt < NT; ++nt)
#pragma unroll
    for (int r = 0; r < 8; ++r)
      Ht[(size_t)(mbase + r) * N + nt * 16 + col] = acc[nt][r];
}

// ---------------- Utility kernels ------------------------------------------
__global__ void fill_kernel(float* __restrict__ p, float v, long n) {
  for (long i = (long)blockIdx.x * blockDim.x + threadIdx.x; i < n;
       i += (long)gridDim.x * blockDim.x)
    p[i] = v;
}

__global__ void cvt_f16_kernel(const float* __restrict__ x,
                               _Float16* __restrict__ y, long n) {
  for (long i = (long)blockIdx.x * blockDim.x + threadIdx.x; i < n;
       i += (long)gridDim.x * blockDim.x)
    y[i] = (_Float16)x[i];
}

__global__ void degree_kernel(const int* __restrict__ dst, float* __restrict__ deg) {
  for (int e = blockIdx.x * blockDim.x + threadIdx.x; e < N_EDGES;
       e += gridDim.x * blockDim.x)
    atomicAdd(&deg[dst[e]], 1.0f);
}

__global__ void dinv_kernel(const float* __restrict__ deg,
                            float* __restrict__ dis, float* __restrict__ dnv) {
  for (int i = blockIdx.x * blockDim.x + threadIdx.x; i < N_NODES;
       i += gridDim.x * blockDim.x) {
    const float d = deg[i];
    dis[i] = rsqrtf(d);
    dnv[i] = 1.0f / d;
  }
}

// agg[dst,f] += h[src,f] * dinv_sqrt[src]*dinv_sqrt[dst]
template <int F>
__global__ void scatter_kernel(const float* __restrict__ h,
                               const int* __restrict__ src,
                               const int* __restrict__ dst,
                               const float* __restrict__ dis,
                               float* __restrict__ agg) {
  const long total = (long)N_EDGES * F;
  for (long i = (long)blockIdx.x * blockDim.x + threadIdx.x; i < total;
       i += (long)gridDim.x * blockDim.x) {
    const int e = (int)(i / F);
    const int f = (int)(i % F);
    const int s = src[e], d = dst[e];
    const float nrm = dis[s] * dis[d];
    atomicAdd(&agg[(size_t)d * F + f], h[(size_t)s * F + f] * nrm);
  }
}

// out = agg + h * dinv[node] + bias[f]   (fp32 out, feeds BN stats)
template <int F>
__global__ void combine_kernel(const float* __restrict__ agg,
                               const float* __restrict__ h,
                               const float* __restrict__ dnv,
                               const float* __restrict__ bias,
                               float* __restrict__ out) {
  const long total = (long)N_NODES * F;
  for (long i = (long)blockIdx.x * blockDim.x + threadIdx.x; i < total;
       i += (long)gridDim.x * blockDim.x) {
    const int node = (int)(i / F);
    const int f    = (int)(i % F);
    out[i] = agg[i] + h[i] * dnv[node] + bias[f];
  }
}

// out = f16(agg + h*dinv + bias)  (final embeddings -> edge MLP)
template <int F>
__global__ void combine_f16_kernel(const float* __restrict__ agg,
                                   const float* __restrict__ h,
                                   const float* __restrict__ dnv,
                                   const float* __restrict__ bias,
                                   _Float16* __restrict__ out) {
  const long total = (long)N_NODES * F;
  for (long i = (long)blockIdx.x * blockDim.x + threadIdx.x; i < total;
       i += (long)gridDim.x * blockDim.x) {
    const int node = (int)(i / F);
    const int f    = (int)(i % F);
    out[i] = (_Float16)(agg[i] + h[i] * dnv[node] + bias[f]);
  }
}

// per-column sum / sum-of-squares with LDS pre-reduction (F divides 256)
template <int F>
__global__ void bn_stats_kernel(const float* __restrict__ x,
                                float* __restrict__ sum, float* __restrict__ sumsq) {
  __shared__ float ssum[256], ssq[256];
  const long total = (long)N_NODES * F;
  float ls = 0.f, lq = 0.f;
  for (long i = (long)blockIdx.x * blockDim.x + threadIdx.x; i < total;
       i += (long)gridDim.x * blockDim.x) {
    const float v = x[i];
    ls += v; lq += v * v;
  }
  ssum[threadIdx.x] = ls; ssq[threadIdx.x] = lq;
  __syncthreads();
#pragma unroll
  for (int off = 128; off >= F; off >>= 1) {
    if ((int)threadIdx.x < off) {
      ssum[threadIdx.x] += ssum[threadIdx.x + off];
      ssq[threadIdx.x]  += ssq[threadIdx.x + off];
    }
    __syncthreads();
  }
  if ((int)threadIdx.x < F) {
    atomicAdd(&sum[threadIdx.x],   ssum[threadIdx.x]);
    atomicAdd(&sumsq[threadIdx.x], ssq[threadIdx.x]);
  }
}

template <int F>
__global__ void bn_final_kernel(const float* __restrict__ sum,
                                const float* __restrict__ sumsq,
                                const float* __restrict__ g,
                                const float* __restrict__ be,
                                float* __restrict__ scale, float* __restrict__ shift) {
  const int f = blockIdx.x * blockDim.x + threadIdx.x;
  if (f < F) {
    const float inv_n = 1.0f / (float)N_NODES;
    const float mu  = sum[f] * inv_n;
    const float var = sumsq[f] * inv_n - mu * mu;
    const float sc  = g[f] * rsqrtf(var + BN_EPS);
    scale[f] = sc;
    shift[f] = be[f] - mu * sc;
  }
}

// BN + ReLU, writing f16 activations for the next WMMA GEMM
template <int F>
__global__ void bn_relu_f16_kernel(const float* __restrict__ x,
                                   const float* __restrict__ scale,
                                   const float* __restrict__ shift,
                                   _Float16* __restrict__ out) {
  const long total = (long)N_NODES * F;
  for (long i = (long)blockIdx.x * blockDim.x + threadIdx.x; i < total;
       i += (long)gridDim.x * blockDim.x) {
    const int f = (int)(i % F);
    const float v = x[i] * scale[f] + shift[f];
    out[i] = (_Float16)(v > 0.f ? v : 0.f);
  }
}

__global__ void relu_f16_kernel(const float* __restrict__ x,
                                _Float16* __restrict__ out, long n) {
  for (long i = (long)blockIdx.x * blockDim.x + threadIdx.x; i < n;
       i += (long)gridDim.x * blockDim.x) {
    const float v = x[i];
    out[i] = (_Float16)(v > 0.f ? v : 0.f);
  }
}

// ---------------- Edge predictor MLP (WMMA) --------------------------------
// One wave per 16-edge tile.  emb is f16 [N_NODES,64]; concat feature k<64
// comes from emb[src] (k-chunks 0,1), k>=64 from emb[dst] (k-chunks 2,3).
#define MLP_WPB 8  // waves per block
__global__ void edge_mlp_kernel(const _Float16* __restrict__ emb,  // [N,64] f16
                                const int* __restrict__ src,
                                const int* __restrict__ dst,
                                const _Float16* __restrict__ WpE1, // packed 128x64
                                const float* __restrict__ bme1,    // [64]
                                const _Float16* __restrict__ WpE2, // packed 64x32
                                const float* __restrict__ bme2,    // [32]
                                const float* __restrict__ We3,     // [32,2]
                                const float* __restrict__ bme3,    // [2]
                                float* __restrict__ out) {         // [E,2]
  __shared__ __align__(16) _Float16 lds1[MLP_WPB][16][64];
  __shared__ float lds2[MLP_WPB][16][32];

  const int wslot = threadIdx.x >> 5;
  const int wave  = blockIdx.x * MLP_WPB + wslot;   // grid sized exactly
  const int lane  = threadIdx.x & 31;
  const int m     = lane & 15;       // edge row within tile
  const int khalf = lane >> 4;
  const int ebase = wave * 16;

  const _Float16* rowS = emb + (size_t)src[ebase + m] * 64;
  const _Float16* rowD = emb + (size_t)dst[ebase + m] * 64;

  // ---- layer 1: (16x128) @ (128x64), 4 K-chunks, 4 N-tiles ----
  v8f acc1[4] = {};
#pragma unroll
  for (int kc = 0; kc < 4; ++kc) {
    const _Float16* row = (kc < 2) ? rowS : rowD;
    const v16h a = load_a_frag_h(row, (kc & 1) * 32, khalf);
#pragma unroll
    for (int nt = 0; nt < 4; ++nt) {
      const v16h b = *(const v16h*)(WpE1 + ((size_t)(kc * 4 + nt) * 32 + lane) * 16);
      acc1[nt] = __builtin_amdgcn_wmma_f32_16x16x32_f16(
          false, a, false, b, (short)0, acc1[nt], false, false);
    }
  }
  {
    const int mbase = khalf * 8;
    const int col0  = lane & 15;
#pragma unroll
    for (int nt = 0; nt < 4; ++nt)
#pragma unroll
      for (int r = 0; r < 8; ++r) {
        const int col = nt * 16 + col0;
        const float v = acc1[nt][r] + bme1[col];
        lds1[wslot][mbase + r][col] = (_Float16)(v > 0.f ? v : 0.f);
      }
  }
  __syncthreads();

  // ---- layer 2: (16x64) @ (64x32), 2 K-chunks, 2 N-tiles ----
  v8f acc2[2] = {};
  const _Float16* a2row = &lds1[wslot][m][0];
#pragma unroll
  for (int kc = 0; kc < 2; ++kc) {
    const v16h a = load_a_frag_h(a2row, kc * 32, khalf);
#pragma unroll
    for (int nt = 0; nt < 2; ++nt) {
      const v16h b = *(const v16h*)(WpE2 + ((size_t)(kc * 2 + nt) * 32 + lane) * 16);
      acc2[nt] = __builtin_amdgcn_wmma_f32_16x16x32_f16(
          false, a, false, b, (short)0, acc2[nt], false, false);
    }
  }
  {
    const int mbase = khalf * 8;
    const int col0  = lane & 15;
#pragma unroll
    for (int nt = 0; nt < 2; ++nt)
#pragma unroll
      for (int r = 0; r < 8; ++r) {
        const int col = nt * 16 + col0;
        const float v = acc2[nt][r] + bme2[col];
        lds2[wslot][mbase + r][col] = v > 0.f ? v : 0.f;
      }
  }
  __syncthreads();

  // ---- layer 3: (16x32) @ (32x2) scalar tail: lane -> (edge, out) pair ----
  {
    const int e_loc = lane >> 1;
    const int o     = lane & 1;
    float acc = bme3[o];
#pragma unroll
    for (int k = 0; k < 32; ++k)
      acc += lds2[wslot][e_loc][k] * We3[k * 2 + o];
    out[(size_t)(ebase + e_loc) * 2 + o] = acc;
  }
}

// ---------------- Launch orchestration -------------------------------------
static inline int nblk(long n, int t) {
  long b = (n + t - 1) / t;
  return (int)(b > 16384 ? 16384 : b);
}

extern "C" void kernel_launch(void* const* d_in, const int* in_sizes, int n_in,
                              void* d_out, int out_size, void* d_ws, size_t ws_size,
                              hipStream_t stream) {
  const float* x    = (const float*)d_in[0];
  const int*   ei   = (const int*)d_in[1];   // [2, E]
  const int*   src  = ei;
  const int*   dstp = ei + N_EDGES;
  const float* W1   = (const float*)d_in[2];
  const float* b1   = (const float*)d_in[3];
  const float* g1   = (const float*)d_in[4];
  const float* be1  = (const float*)d_in[5];
  const float* W2   = (const float*)d_in[6];
  const float* b2   = (const float*)d_in[7];
  const float* g2   = (const float*)d_in[8];
  const float* be2  = (const float*)d_in[9];
  const float* W3   = (const float*)d_in[10];
  const float* b3   = (const float*)d_in[11];
  const float* We1  = (const float*)d_in[12];
  const float* bme1 = (const float*)d_in[13];
  const float* We2  = (const float*)d_in[14];
  const float* bme2 = (const float*)d_in[15];
  const float* We3  = (const float*)d_in[16];
  const float* bme3 = (const float*)d_in[17];
  float* out = (float*)d_out;

  // workspace carve-up
  float* ws = (float*)d_ws;
  const size_t NF = (size_t)N_NODES * 256;
  float* A      = ws;            // fp32 node buffer A  [50000*256]
  float* B      = A + NF;        // fp32 node buffer B
  float* C      = B + NF;        // fp32 node buffer C
  float* deg    = C + NF;        // [50000]
  float* dis    = deg + N_NODES; // deg^-1/2
  float* dnv    = dis + N_NODES; // deg^-1
  float* bsum   = dnv + N_NODES; // [256]
  float* bsq    = bsum + 256;
  float* bscale = bsq + 256;
  float* bshift = bscale + 256;
  _Float16* XH1 = (_Float16*)(bshift + 256);          // f16 activations [50000*256]
  _Float16* XH2 = XH1 + (size_t)N_NODES * 256;        // f16 activations [50000*64]
  _Float16* EH  = XH2 + (size_t)N_NODES * 64;         // f16 embeddings  [50000*64]
  _Float16* Wp1 = EH  + (size_t)N_NODES * 64;         // packed 256x256
  _Float16* Wp2 = Wp1 + 256 * 256;                    // packed 256x64
  _Float16* Wp3 = Wp2 + 256 * 64;                     // packed 64x64
  _Float16* WpE1 = Wp3 + 64 * 64;                     // packed 128x64
  _Float16* WpE2 = WpE1 + 128 * 64;                   // packed 64x32

  const int T = 256;
  const int gemm_blocks = (MTILES + 7) / 8;   // 8 waves/block

  // ---- one-time per call: pack weights, convert x, degrees ----
  pack_b_kernel<<<nblk(256 * 16 * 32, T), T, 0, stream>>>(W1, Wp1, 256, 256);
  pack_b_kernel<<<nblk(8 * 4 * 32, T), T, 0, stream>>>(W2, Wp2, 256, 64);
  pack_b_kernel<<<nblk(2 * 4 * 32, T), T, 0, stream>>>(W3, Wp3, 64, 64);
  pack_b_kernel<<<nblk(4 * 4 * 32, T), T, 0, stream>>>(We1, WpE1, 128, 64);
  pack_b_kernel<<<nblk(2 * 2 * 32, T), T, 0, stream>>>(We2, WpE2, 64, 32);
  cvt_f16_kernel<<<nblk((long)N_NODES * 256, T), T, 0, stream>>>(x, XH1, (long)N_NODES * 256);

  fill_kernel<<<nblk(N_NODES, T), T, 0, stream>>>(deg, 1.0f, N_NODES);
  degree_kernel<<<nblk(N_EDGES, T), T, 0, stream>>>(dstp, deg);
  dinv_kernel<<<nblk(N_NODES, T), T, 0, stream>>>(deg, dis, dnv);

  // ---- Layer 1: 256 -> 256, BN, ReLU ----
  gemm_wmma_kernel<256, 256><<<gemm_blocks, T, 0, stream>>>(XH1, Wp1, A, MTILES);
  fill_kernel<<<nblk((long)N_NODES * 256, T), T, 0, stream>>>(B, 0.f, (long)N_NODES * 256);
  scatter_kernel<256><<<nblk((long)N_EDGES * 256, T), T, 0, stream>>>(A, src, dstp, dis, B);
  combine_kernel<256><<<nblk((long)N_NODES * 256, T), T, 0, stream>>>(B, A, dnv, b1, C);
  fill_kernel<<<1, 512, 0, stream>>>(bsum, 0.f, 512);
  bn_stats_kernel<256><<<nblk((long)N_NODES * 256, T), T, 0, stream>>>(C, bsum, bsq);
  bn_final_kernel<256><<<1, 256, 0, stream>>>(bsum, bsq, g1, be1, bscale, bshift);
  bn_relu_f16_kernel<256><<<nblk((long)N_NODES * 256, T), T, 0, stream>>>(C, bscale, bshift, XH1);

  // ---- Layer 2: 256 -> 64, BN, ReLU ----
  gemm_wmma_kernel<256, 64><<<gemm_blocks, T, 0, stream>>>(XH1, Wp2, A, MTILES);
  fill_kernel<<<nblk((long)N_NODES * 64, T), T, 0, stream>>>(C, 0.f, (long)N_NODES * 64);
  scatter_kernel<64><<<nblk((long)N_EDGES * 64, T), T, 0, stream>>>(A, src, dstp, dis, C);
  combine_kernel<64><<<nblk((long)N_NODES * 64, T), T, 0, stream>>>(C, A, dnv, b2, C);
  fill_kernel<<<1, 512, 0, stream>>>(bsum, 0.f, 512);
  bn_stats_kernel<64><<<nblk((long)N_NODES * 64, T), T, 0, stream>>>(C, bsum, bsq);
  bn_final_kernel<64><<<1, 64, 0, stream>>>(bsum, bsq, g2, be2, bscale, bshift);
  bn_relu_f16_kernel<64><<<nblk((long)N_NODES * 64, T), T, 0, stream>>>(C, bscale, bshift, XH2);

  // ---- Layer 3: 64 -> 64, ReLU ----
  gemm_wmma_kernel<64, 64><<<gemm_blocks, T, 0, stream>>>(XH2, Wp3, B, MTILES);
  fill_kernel<<<nblk((long)N_NODES * 64, T), T, 0, stream>>>(C, 0.f, (long)N_NODES * 64);
  scatter_kernel<64><<<nblk((long)N_EDGES * 64, T), T, 0, stream>>>(B, src, dstp, dis, C);
  combine_kernel<64><<<nblk((long)N_NODES * 64, T), T, 0, stream>>>(C, B, dnv, b3, C);
  relu_f16_kernel<<<nblk((long)N_NODES * 64, T), T, 0, stream>>>(C, XH2, (long)N_NODES * 64);

  // ---- Layer 4: 64 -> 64 (same W3/b3, no ReLU) -> f16 embeddings ----
  gemm_wmma_kernel<64, 64><<<gemm_blocks, T, 0, stream>>>(XH2, Wp3, A, MTILES);
  fill_kernel<<<nblk((long)N_NODES * 64, T), T, 0, stream>>>(B, 0.f, (long)N_NODES * 64);
  scatter_kernel<64><<<nblk((long)N_EDGES * 64, T), T, 0, stream>>>(A, src, dstp, dis, B);
  combine_f16_kernel<64><<<nblk((long)N_NODES * 64, T), T, 0, stream>>>(B, A, dnv, b3, EH);

  // ---- Edge MLP: f16 gather + 128->64->32->2 ----
  edge_mlp_kernel<<<ETILES / MLP_WPB, MLP_WPB * 32, 0, stream>>>(
      EH, src, dstp, WpE1, bme1, WpE2, bme2, We3, bme3, out);
}
